// GCN_5360119185947
// MI455X (gfx1250) — matrix operand order repaired and checked by reading, not compiled, not run on hardware
//
#include <hip/hip_runtime.h>
#include <hip/hip_bf16.h>

#define N_NODES 100000
#define N_EDGES 1600000
#define F_IN    128
#define HID     256
#define H2      196
#define DOUT    128
#define NB      512
#define BN_EPS  1e-5f

typedef __attribute__((ext_vector_type(16))) __bf16 v16bf;
typedef __attribute__((ext_vector_type(8)))  float  v8f;

union V16U { uint4 u[2]; v16bf v; };

// ---------------- elementwise / setup kernels ----------------

__global__ void k_set_one(float* p, int n) {
  int i = blockIdx.x * blockDim.x + threadIdx.x;
  if (i < n) p[i] = 1.0f;
}

__global__ void k_deg_count(const long long* __restrict__ dst, float* __restrict__ deg, int ne) {
  int e = blockIdx.x * blockDim.x + threadIdx.x;
  if (e < ne) atomicAdd(&deg[(int)dst[e]], 1.0f);
}

__global__ void k_rsqrt_inplace(float* p, int n) {
  int i = blockIdx.x * blockDim.x + threadIdx.x;
  if (i < n) p[i] = rsqrtf(p[i]);
}

__global__ void k_cast_bf16(const float* __restrict__ in, __bf16* __restrict__ out, size_t n) {
  size_t i = (size_t)blockIdx.x * blockDim.x + threadIdx.x;
  if (i < n) out[i] = (__bf16)in[i];
}

// W[K][Nc] (row major) -> Wt[Nc][K] bf16
__global__ void k_transpose_bf16(const float* __restrict__ W, __bf16* __restrict__ Wt, int K, int Nc) {
  int i = blockIdx.x * blockDim.x + threadIdx.x;
  if (i >= K * Nc) return;
  int k = i / Nc, n = i % Nc;
  Wt[(size_t)n * K + k] = (__bf16)W[i];
}

// ---------------- WMMA GEMM: C[M,Nc] = A[M,K] @ B[K,Nc], B given transposed ----------------
// One wave computes a 16x16 tile, K-step 32 via v_wmma_f32_16x16x32_bf16.

__global__ __launch_bounds__(256) void k_wmma_gemm(
    const __bf16* __restrict__ A,   // [M,K] row-major bf16
    const __bf16* __restrict__ Bt,  // [Nc,K] row-major bf16 (= B transposed)
    float* __restrict__ C,          // [M,Nc] f32
    int M, int K, int Nc)
{
  const int wpb   = blockDim.x >> 5;
  const int wave  = threadIdx.x >> 5;
  const int lane  = threadIdx.x & 31;
  const int tilesN = Nc >> 4;
  int tile = blockIdx.x * wpb + wave;
  int total = (M >> 4) * tilesN;
  if (tile >= total) return;
  int tm = tile / tilesN, tn = tile - tm * tilesN;
  int m0 = tm << 4, n0 = tn << 4;
  int half = lane >> 4;
  int l15  = lane & 15;

  const __bf16* arow = A  + (size_t)(m0 + l15) * K;
  const __bf16* brow = Bt + (size_t)(n0 + l15) * K;

  v8f acc = {};
  for (int k0 = 0; k0 < K; k0 += 32) {
    V16U af, bf;
    // A 16x32 bf16 layout: lane half 0 -> K {0..7,16..23}, half 1 -> K {8..15,24..31}
    af.u[0] = *(const uint4*)(arow + k0 + 8 * half);
    af.u[1] = *(const uint4*)(arow + k0 + 16 + 8 * half);
    // B 32x16 bf16 layout: lane half 0 -> K {0..7,16..23}, half 1 -> K {8..15,24..31}
    bf.u[0] = *(const uint4*)(brow + k0 + 8 * half);
    bf.u[1] = *(const uint4*)(brow + k0 + 16 + 8 * half);
    acc = __builtin_amdgcn_wmma_f32_16x16x32_bf16(
        false, af.v, false, bf.v, (short)0, acc, false, false);
  }
  // C/D layout: VGPR r -> row m0 + r + 8*half, col n0 + l15
  float* crow = C + (size_t)(m0 + 8 * half) * Nc + n0 + l15;
#pragma unroll
  for (int r = 0; r < 8; ++r)
    crow[(size_t)r * Nc] = acc[r];
}

// ---------------- edge aggregation: agg[dst] += hh[src] * dis[src]*dis[dst] ----------------

__global__ __launch_bounds__(256) void k_edge_agg(
    const float* __restrict__ hh, const long long* __restrict__ src,
    const long long* __restrict__ dst, const float* __restrict__ dis,
    float* __restrict__ agg, int ne)
{
  int wid  = blockIdx.x * (blockDim.x >> 5) + (threadIdx.x >> 5);
  int lane = threadIdx.x & 31;
  int nw   = gridDim.x * (blockDim.x >> 5);
  for (int e = wid; e < ne; e += nw) {
    int s = (int)src[e], d = (int)dst[e];
    float w = dis[s] * dis[d];
    const float4* hs = (const float4*)(hh + (size_t)s * HID);
    float* ad = agg + (size_t)d * HID + lane * 8;
    float4 v0 = hs[lane * 2 + 0];
    float4 v1 = hs[lane * 2 + 1];
    atomicAdd(ad + 0, v0.x * w); atomicAdd(ad + 1, v0.y * w);
    atomicAdd(ad + 2, v0.z * w); atomicAdd(ad + 3, v0.w * w);
    atomicAdd(ad + 4, v1.x * w); atomicAdd(ad + 5, v1.y * w);
    atomicAdd(ad + 6, v1.z * w); atomicAdd(ad + 7, v1.w * w);
  }
}

// ---------------- self-loop + bias + BN stats (fused) ----------------
// blockDim.x == HID; thread = channel, rows block-strided (coalesced)

__global__ __launch_bounds__(HID) void k_finalize_conv(
    float* __restrict__ agg, const float* __restrict__ hh,
    const float* __restrict__ dis, const float* __restrict__ bias,
    float* __restrict__ gsum, float* __restrict__ gsumsq, int rows)
{
  int c = threadIdx.x;
  float b = bias[c];
  float s = 0.f, sq = 0.f;
  for (int i = blockIdx.x; i < rows; i += gridDim.x) {
    float dv = dis[i];
    size_t off = (size_t)i * HID + c;
    float v = agg[off] + hh[off] * (dv * dv) + b;
    agg[off] = v;
    s += v; sq += v * v;
  }
  atomicAdd(&gsum[c], s);
  atomicAdd(&gsumsq[c], sq);
}

// generic column stats for the head BN
__global__ __launch_bounds__(256) void k_col_stats(
    const float* __restrict__ X, float* __restrict__ gsum, float* __restrict__ gsumsq,
    int rows, int cols)
{
  int c = threadIdx.x;
  if (c >= cols) return;
  float s = 0.f, sq = 0.f;
  for (int r = blockIdx.x; r < rows; r += gridDim.x) {
    float v = X[(size_t)r * cols + c];
    s += v; sq += v * v;
  }
  atomicAdd(&gsum[c], s);
  atomicAdd(&gsumsq[c], sq);
}

__global__ void k_bn_finalize(const float* __restrict__ gsum, const float* __restrict__ gsumsq,
                              float* __restrict__ mu, float* __restrict__ rstd,
                              int cols, float invR)
{
  int c = blockIdx.x * blockDim.x + threadIdx.x;
  if (c >= cols) return;
  float m = gsum[c] * invR;
  float v = gsumsq[c] * invR - m * m;
  mu[c] = m;
  rstd[c] = rsqrtf(v + BN_EPS);
}

__global__ void k_bn_apply_relu(const float* __restrict__ X, float* __restrict__ Y32,
                                __bf16* __restrict__ Ybf,
                                const float* __restrict__ mu, const float* __restrict__ rstd,
                                const float* __restrict__ g, const float* __restrict__ be,
                                size_t n, int cols)
{
  size_t idx = (size_t)blockIdx.x * blockDim.x + threadIdx.x;
  if (idx >= n) return;
  int c = (int)(idx % cols);
  float v = (X[idx] - mu[c]) * rstd[c] * g[c] + be[c];
  v = fmaxf(v, 0.f);
  if (Y32) Y32[idx] = v;
  if (Ybf) Ybf[idx] = (__bf16)v;
}

// ---------------- pooling ----------------

__global__ void k_count_nodes(const long long* __restrict__ batch, float* __restrict__ counts, int n) {
  int i = blockIdx.x * blockDim.x + threadIdx.x;
  if (i < n) atomicAdd(&counts[(int)batch[i]], 1.0f);
}

__global__ void k_pool_sum(const float* __restrict__ h, const long long* __restrict__ batch,
                           float* __restrict__ hg_sum, size_t n, int cols)
{
  size_t idx = (size_t)blockIdx.x * blockDim.x + threadIdx.x;
  if (idx >= n) return;
  size_t i = idx / cols;
  int   c = (int)(idx - i * cols);
  int   b = (int)batch[i];
  atomicAdd(&hg_sum[(size_t)b * cols + c], h[idx]);
}

__global__ void k_hg_final(const float* __restrict__ hg_sum, const float* __restrict__ counts,
                           float* __restrict__ hg, int n, int cols)
{
  int idx = blockIdx.x * blockDim.x + threadIdx.x;
  if (idx >= n) return;
  int b = idx / cols;
  hg[idx] = hg_sum[idx] / fmaxf(counts[b], 1.0f);
}

// ---------------- small head GEMM: one block per output row, X-row staged in LDS ----------------

__global__ __launch_bounds__(256) void k_head_gemm(
    const float* __restrict__ X, const float* __restrict__ W, const float* __restrict__ bias,
    float* __restrict__ Y, int K, int Ncols)
{
  __shared__ float s[256];
  int b = blockIdx.x;
  for (int k = threadIdx.x; k < K; k += blockDim.x)
    s[k] = X[(size_t)b * K + k];
  __syncthreads();
  for (int j = threadIdx.x; j < Ncols; j += blockDim.x) {
    float acc = bias[j];
    for (int k = 0; k < K; ++k)
      acc += s[k] * W[(size_t)k * Ncols + j];   // coalesced over j
    Y[(size_t)b * Ncols + j] = acc;
  }
}

// ---------------- launch ----------------

extern "C" void kernel_launch(void* const* d_in, const int* in_sizes, int n_in,
                              void* d_out, int out_size, void* d_ws, size_t ws_size,
                              hipStream_t stream)
{
  const float*      x     = (const float*)d_in[0];
  const long long*  ei    = (const long long*)d_in[1];
  const long long*  batch = (const long long*)d_in[2];
  const float* W1  = (const float*)d_in[3];
  const float* b1  = (const float*)d_in[4];
  const float* g1  = (const float*)d_in[5];
  const float* be1 = (const float*)d_in[6];
  const float* W2  = (const float*)d_in[7];
  const float* b2  = (const float*)d_in[8];
  const float* g2  = (const float*)d_in[9];
  const float* be2 = (const float*)d_in[10];
  const float* Wf2 = (const float*)d_in[11];
  const float* bf2 = (const float*)d_in[12];
  const float* g3  = (const float*)d_in[13];
  const float* be3 = (const float*)d_in[14];
  const float* Wf3 = (const float*)d_in[15];
  const float* bf3 = (const float*)d_in[16];

  const long long* esrc = ei;
  const long long* edst = ei + N_EDGES;

  char* ws = (char*)d_ws;
  size_t off = 0;
  auto take = [&](size_t bytes) -> char* {
    char* p = ws + off;
    off = (off + bytes + 255) & ~(size_t)255;
    return p;
  };
  float*  hh     = (float*)take((size_t)N_NODES * HID * 4);   // GEMM out; later act f32
  float*  agg    = (float*)take((size_t)N_NODES * HID * 4);   // aggregation / conv out
  __bf16* bfb    = (__bf16*)take((size_t)N_NODES * HID * 2);  // x_bf16 then act1_bf16
  float*  dis    = (float*)take((size_t)N_NODES * 4);         // deg -> rsqrt(deg)
  __bf16* W1t    = (__bf16*)take((size_t)HID * F_IN * 2);
  __bf16* W2t    = (__bf16*)take((size_t)HID * HID * 2);
  float*  stats  = (float*)take(1024 * 4);                    // gsum|gsumsq|mu|rstd
  float*  hg_sum = (float*)take((size_t)NB * HID * 4);
  float*  counts = (float*)take((size_t)NB * 4);
  float*  z1     = (float*)take((size_t)NB * H2 * 4);
  float*  h2buf  = (float*)take((size_t)NB * H2 * 4);

  float* gsum   = stats;
  float* gsumsq = stats + 256;
  float* mu     = stats + 512;
  float* rstd   = stats + 768;

  float* out = (float*)d_out;                   // [512,128]
  float* hg  = out + (size_t)NB * DOUT;         // [512,256] (2nd tuple output)

  const int T = 256;
  auto cdiv = [](size_t a, size_t b) -> int { return (int)((a + b - 1) / b); };

  // degree / norm
  k_set_one<<<cdiv(N_NODES, T), T, 0, stream>>>(dis, N_NODES);
  k_deg_count<<<cdiv(N_EDGES, T), T, 0, stream>>>(edst, dis, N_EDGES);
  k_rsqrt_inplace<<<cdiv(N_NODES, T), T, 0, stream>>>(dis, N_NODES);

  // bf16 operands
  k_cast_bf16<<<cdiv((size_t)N_NODES * F_IN, T), T, 0, stream>>>(x, bfb, (size_t)N_NODES * F_IN);
  k_transpose_bf16<<<cdiv((size_t)F_IN * HID, T), T, 0, stream>>>(W1, W1t, F_IN, HID);
  k_transpose_bf16<<<cdiv((size_t)HID * HID, T), T, 0, stream>>>(W2, W2t, HID, HID);

  const int gemm_blocks = (N_NODES / 16) * (HID / 16) / 8;   // 12500

  // ---- layer 1 ----
  k_wmma_gemm<<<gemm_blocks, T, 0, stream>>>(bfb, W1t, hh, N_NODES, F_IN, HID);
  hipMemsetAsync(agg, 0, (size_t)N_NODES * HID * 4, stream);
  hipMemsetAsync(stats, 0, 512 * 4, stream);
  k_edge_agg<<<8192, T, 0, stream>>>(hh, esrc, edst, dis, agg, N_EDGES);
  k_finalize_conv<<<2048, HID, 0, stream>>>(agg, hh, dis, b1, gsum, gsumsq, N_NODES);
  k_bn_finalize<<<1, T, 0, stream>>>(gsum, gsumsq, mu, rstd, HID, 1.0f / N_NODES);
  k_bn_apply_relu<<<cdiv((size_t)N_NODES * HID, T), T, 0, stream>>>(
      agg, (float*)nullptr, bfb, mu, rstd, g1, be1, (size_t)N_NODES * HID, HID);

  // ---- layer 2 ----
  k_wmma_gemm<<<gemm_blocks, T, 0, stream>>>(bfb, W2t, hh, N_NODES, HID, HID);
  hipMemsetAsync(agg, 0, (size_t)N_NODES * HID * 4, stream);
  hipMemsetAsync(stats, 0, 512 * 4, stream);
  k_edge_agg<<<8192, T, 0, stream>>>(hh, esrc, edst, dis, agg, N_EDGES);
  k_finalize_conv<<<2048, HID, 0, stream>>>(agg, hh, dis, b2, gsum, gsumsq, N_NODES);
  k_bn_finalize<<<1, T, 0, stream>>>(gsum, gsumsq, mu, rstd, HID, 1.0f / N_NODES);
  k_bn_apply_relu<<<cdiv((size_t)N_NODES * HID, T), T, 0, stream>>>(
      agg, hh, (__bf16*)nullptr, mu, rstd, g2, be2, (size_t)N_NODES * HID, HID);

  // ---- global mean pool ----
  hipMemsetAsync(hg_sum, 0, (size_t)NB * HID * 4, stream);
  hipMemsetAsync(counts, 0, (size_t)NB * 4, stream);
  k_count_nodes<<<cdiv(N_NODES, T), T, 0, stream>>>(batch, counts, N_NODES);
  k_pool_sum<<<cdiv((size_t)N_NODES * HID, T), T, 0, stream>>>(
      hh, batch, hg_sum, (size_t)N_NODES * HID, HID);
  k_hg_final<<<cdiv((size_t)NB * HID, T), T, 0, stream>>>(hg_sum, counts, hg, NB * HID, HID);

  // ---- MLP head ----
  k_head_gemm<<<NB, T, 0, stream>>>(hg, Wf2, bf2, z1, HID, H2);
  hipMemsetAsync(stats, 0, 512 * 4, stream);
  k_col_stats<<<256, T, 0, stream>>>(z1, gsum, gsumsq, NB, H2);
  k_bn_finalize<<<1, T, 0, stream>>>(gsum, gsumsq, mu, rstd, H2, 1.0f / NB);
  k_bn_apply_relu<<<cdiv((size_t)NB * H2, T), T, 0, stream>>>(
      z1, h2buf, (__bf16*)nullptr, mu, rstd, g3, be3, (size_t)NB * H2, H2);
  k_head_gemm<<<NB, T, 0, stream>>>(h2buf, Wf3, bf3, out, H2, DOUT);
}
